// _VisualFieldAttentionModule_4020089389091
// MI455X (gfx1250) — matrix operand order repaired and checked by reading, not compiled, not run on hardware
//
#include <hip/hip_runtime.h>

// ---------------------------------------------------------------------------
// VisualFieldAttentionModule on gfx1250 (MI455X), bf16 WMMA flash-attention.
// B=4, C=512, CQK=64, H=W=64, N=4096.
//
// k_prep : weights f32 -> bf16 [o][c] (B-operand friendly)
// k_qkv  : fused QKV projection per 16-pixel tile (WMMA over K=512)
// k_attn : per (batch, 16-query tile): scores -> LDS, exact 2-pass softmax
//          with in-place f32 -> packed-bf16 conversion, then P @ V^T with
//          fused (alpha * O / rowsum + x) epilogue.
// ---------------------------------------------------------------------------

#define BATCH 4
#define CIN   512
#define CQK   64
#define NPIX  4096
#define SROW  4100   // padded LDS row stride (dwords) for the score buffer

typedef __attribute__((ext_vector_type(16))) __bf16 v16bf;
typedef __attribute__((ext_vector_type(8)))  float  v8f;

union ABFrag {
    v16bf          v;
    unsigned short h[16];
    unsigned       d[8];
    uint4          q[2];
};

__device__ __forceinline__ unsigned short f2bf(float f) {
    union { float f; unsigned u; } a; a.f = f;
    unsigned u = a.u;
    unsigned r = (u + 0x7FFFu + ((u >> 16) & 1u)) >> 16;   // round-to-nearest-even
    return (unsigned short)r;
}

__device__ __forceinline__ v8f wmma_bf16(const ABFrag& a, const ABFrag& b, v8f c) {
    return __builtin_amdgcn_wmma_f32_16x16x32_bf16(
        /*neg_a=*/false, a.v, /*neg_b=*/false, b.v,
        /*c_mod=*/(short)0, c, /*reuse_a=*/false, /*reuse_b=*/false);
}

// ---------------------------------------------------------------------------
// Kernel 0: convert weights f32 -> bf16, row-major [o][c]
// ---------------------------------------------------------------------------
__global__ void k_prep(const float* __restrict__ wb, const float* __restrict__ wc,
                       const float* __restrict__ wd,
                       unsigned short* __restrict__ wb_bf,
                       unsigned short* __restrict__ wc_bf,
                       unsigned short* __restrict__ wd_bf) {
    const int NB = CQK * CIN;   // 32768
    const int ND = CIN * CIN;   // 262144
    int i = blockIdx.x * blockDim.x + threadIdx.x;
    if (i < NB)               wb_bf[i]          = f2bf(wb[i]);
    else if (i < 2 * NB)      wc_bf[i - NB]     = f2bf(wc[i - NB]);
    else if (i < 2 * NB + ND) wd_bf[i - 2 * NB] = f2bf(wd[i - 2 * NB]);
}

// ---------------------------------------------------------------------------
// Kernel 1: fused QKV projection (1x1 convs) for one 16-pixel tile per block.
//   Q[b][n][o] = lf[n] * (sum_c x[b][c][n] * wb[o][c] + bb[o])   (bf16)
//   K[b][n][o] =          sum_c x[b][c][n] * wc[o][c] + bc[o]    (bf16)
//   V[b][o][n] =          sum_c x[b][c][n] * wd[o][c] + bd[o]    (bf16, transposed)
// ---------------------------------------------------------------------------
__launch_bounds__(256, 2)
__global__ void k_qkv(const float* __restrict__ x, const float* __restrict__ lf,
                      const unsigned short* __restrict__ wb_bf, const float* __restrict__ bb,
                      const unsigned short* __restrict__ wc_bf, const float* __restrict__ bc,
                      const unsigned short* __restrict__ wd_bf, const float* __restrict__ bd,
                      unsigned short* __restrict__ Qbf,
                      unsigned short* __restrict__ Kbf,
                      unsigned short* __restrict__ Vbf) {
    __shared__ unsigned short Xs[16 * 520];   // [j][c], padded rows (16.6 KB)

    const int tile = blockIdx.x & 255;
    const int b    = blockIdx.x >> 8;
    const int n0   = tile * 16;
    const int t    = threadIdx.x;
    const float* xb = x + (size_t)b * CIN * NPIX;

    // stage X^T tile into LDS as bf16: Xs[j][c] = x[b][c][n0+j]
    for (int idx = t; idx < CIN * 16; idx += 256) {
        int c = idx >> 4;
        int j = idx & 15;
        Xs[j * 520 + c] = f2bf(xb[(size_t)c * NPIX + n0 + j]);
    }
    __syncthreads();

    const int wave = t >> 5;
    const int lane = t & 31;
    const int half = lane >> 4;
    const int col  = lane & 15;

    for (int chunk = wave; chunk < 40; chunk += 8) {
        const unsigned short* W; const float* bias; int o0; int kind;
        if (chunk < 4)      { W = wb_bf; bias = bb; o0 = chunk * 16;        kind = 0; }
        else if (chunk < 8) { W = wc_bf; bias = bc; o0 = (chunk - 4) * 16;  kind = 1; }
        else                { W = wd_bf; bias = bd; o0 = (chunk - 8) * 16;  kind = 2; }

        v8f acc = {};
        const unsigned short* arow = &Xs[col * 520];
        for (int ck = 0; ck < CIN; ck += 32) {
            ABFrag a, bf;
            // A fragment: row M = lane%16, per-lane contiguous K (+half*16)
#pragma unroll
            for (int r = 0; r < 8; ++r) {
                int k0 = ck + (half << 4) + (r << 1);
                a.d[r] = *(const unsigned*)(arow + k0);
            }
            // B fragment: col N = o0+lane%16, per-lane contiguous K (+half*16)
            const uint4* wp = (const uint4*)(W + (size_t)(o0 + col) * CIN + ck + (half << 4));
            bf.q[0] = wp[0];
            bf.q[1] = wp[1];
            acc = wmma_bf16(a, bf, acc);
        }

        // epilogue: D element (M = r + 8*half, N = o0 + col)
        const int   o  = o0 + col;
        const float bv = bias[o];
#pragma unroll
        for (int r = 0; r < 8; ++r) {
            int   n = n0 + r + (half << 3);
            float v = acc[r] + bv;
            if (kind == 0) {
                v *= lf[n];
                Qbf[((size_t)b * NPIX + n) * CQK + o] = f2bf(v);
            } else if (kind == 1) {
                Kbf[((size_t)b * NPIX + n) * CQK + o] = f2bf(v);
            } else {
                Vbf[((size_t)b * CIN + o) * NPIX + n] = f2bf(v);
            }
        }
    }
}

// ---------------------------------------------------------------------------
// Kernel 2: attention for one (batch, 16-query tile) per block.
// ---------------------------------------------------------------------------
__launch_bounds__(256, 1)
__global__ void k_attn(const unsigned short* __restrict__ Qbf,
                       const unsigned short* __restrict__ Kbf,
                       const unsigned short* __restrict__ Vbf,
                       const float* __restrict__ x,
                       const float* __restrict__ alpha_p,
                       float* __restrict__ out) {
    extern __shared__ float smem[];
    float*    S       = smem;                 // 16 * SROW f32 scores (later packed bf16)
    unsigned* Spack   = (unsigned*)smem;      // same memory, packed-bf16 view
    float*    red     = smem + 16 * SROW;     // 16x16 partial reductions
    float*    rowstat = red + 256;            // [0..15]=rowmax, [16..31]=rowsum

    const int tile = blockIdx.x & 255;
    const int b    = blockIdx.x >> 8;
    const int n0   = tile * 16;
    const int t    = threadIdx.x;
    const int wave = t >> 5;
    const int lane = t & 31;
    const int half = lane >> 4;
    const int col  = lane & 15;

    // ---- Phase 1: S[16][4096] = Q(16x64) @ K^T, f32 into LDS ----
    {
        ABFrag qa[2];
        const unsigned short* qrow = Qbf + ((size_t)b * NPIX + n0 + col) * CQK;
#pragma unroll
        for (int ck = 0; ck < 2; ++ck)
#pragma unroll
            for (int r = 0; r < 8; ++r) {
                int k0 = ck * 32 + (half << 4) + (r << 1);
                qa[ck].d[r] = *(const unsigned*)(qrow + k0);
            }

        for (int kt = wave; kt < 256; kt += 8) {
            const int m0 = kt << 4;
            const unsigned short* krow =
                Kbf + ((size_t)b * NPIX + m0 + col) * CQK + (half << 4);
            ABFrag kb0, kb1;
            const uint4* kp0 = (const uint4*)krow;
            const uint4* kp1 = (const uint4*)(krow + 32);
            kb0.q[0] = kp0[0]; kb0.q[1] = kp0[1];
            kb1.q[0] = kp1[0]; kb1.q[1] = kp1[1];

            v8f s = {};
            s = wmma_bf16(qa[0], kb0, s);
            s = wmma_bf16(qa[1], kb1, s);
#pragma unroll
            for (int r = 0; r < 8; ++r)
                S[(r + (half << 3)) * SROW + m0 + col] = s[r];
        }
    }
    __syncthreads();

    // ---- Phase 2: rowmax, then exp + rowsum + IN-PLACE bf16 packing ----
    // Thread t owns row t%16, contiguous chunk of 256 m-values. In the exp
    // pass we write packed dword j after consuming f32 elements 2j, 2j+1
    // (forward order => safe in place). After this phase, row n of P lives at
    // Spack[n*SROW + (m/256)*256 + (m%256)/2] as packed bf16 pairs.
    {
        const int row = t & 15;
        const int ch  = t >> 4;
        float*    p  = S     + row * SROW + ch * 256;
        unsigned* pp = Spack + row * SROW + ch * 256;

        float mx = -3.4e38f;
        for (int i = 0; i < 256; ++i) mx = fmaxf(mx, p[i]);
        red[row * 16 + ch] = mx;
        __syncthreads();
        if (t < 16) {
            float m = red[t * 16];
            for (int i = 1; i < 16; ++i) m = fmaxf(m, red[t * 16 + i]);
            rowstat[t] = m;
        }
        __syncthreads();

        const float m = rowstat[row];
        float sum = 0.f;
        for (int i = 0; i < 256; i += 2) {
            float e0 = __expf(p[i]     - m);
            float e1 = __expf(p[i + 1] - m);
            sum += e0 + e1;
            pp[i >> 1] = (unsigned)f2bf(e0) | ((unsigned)f2bf(e1) << 16);
        }
        red[row * 16 + ch] = sum;
        __syncthreads();
        if (t < 16) {
            float s = 0.f;
            for (int i = 0; i < 16; ++i) s += red[t * 16 + i];
            rowstat[16 + t] = s;
        }
    }
    __syncthreads();

    // ---- Phase 3: O = P @ V^T (wave owns channels [c0, c0+64)) ----
    // Packed-bf16 rows: element pair {k0,k0+1} of row n sits at dword
    // Spack[n*SROW + ch_base + k0/2] where ch_base folds the chunked packing:
    // chunk ch packed its 256 elements [ch*256, ch*256+256) into dwords
    // [ch*256, ch*256+128). So global m pair (m0+k0) lives at
    // dword (m/256)*256 + ((m%256)>>1).
    {
        const int c0 = wave * 64;
        v8f acc[4] = {};
        const unsigned* prow = Spack + col * SROW;
        const unsigned short* vbase = Vbf + ((size_t)b * CIN) * NPIX;

        for (int m0 = 0; m0 < NPIX; m0 += 32) {
            const int mch  = (m0 >> 8) << 8;        // chunk base (multiple of 256)
            const int mrem = m0 & 255;              // offset within chunk (mult of 32)
            const unsigned* pchunk = prow + mch + (mrem >> 1);
            ABFrag a;
#pragma unroll
            for (int r = 0; r < 8; ++r) {
                int k0 = (half << 4) + (r << 1);
                a.d[r] = pchunk[k0 >> 1];
            }
            if (m0 + 32 < NPIX)
                __builtin_prefetch(vbase + (size_t)(c0 + col) * NPIX + m0 + 32, 0, 1);
#pragma unroll
            for (int cc = 0; cc < 4; ++cc) {
                const uint4* vp = (const uint4*)(
                    vbase + (size_t)(c0 + cc * 16 + col) * NPIX + m0 + (half << 4));
                ABFrag vb;
                vb.q[0] = vp[0];
                vb.q[1] = vp[1];
                acc[cc] = wmma_bf16(a, vb, acc[cc]);
            }
        }

        // fused epilogue: out = alpha * O / rowsum + x
        const float alpha = alpha_p[0];
#pragma unroll
        for (int cc = 0; cc < 4; ++cc) {
            const int c = c0 + cc * 16 + col;
            const float* xp = x   + ((size_t)b * CIN + c) * NPIX;
            float*       op = out + ((size_t)b * CIN + c) * NPIX;
#pragma unroll
            for (int r = 0; r < 8; ++r) {
                int   nl  = r + (half << 3);
                int   n   = n0 + nl;
                float val = acc[cc][r] / rowstat[16 + nl];
                op[n] = alpha * val + xp[n];
            }
        }
    }
}

// ---------------------------------------------------------------------------
// Host launcher
// ---------------------------------------------------------------------------
extern "C" void kernel_launch(void* const* d_in, const int* in_sizes, int n_in,
                              void* d_out, int out_size, void* d_ws, size_t ws_size,
                              hipStream_t stream) {
    const float* x     = (const float*)d_in[0];
    const float* lf    = (const float*)d_in[1];
    const float* wb    = (const float*)d_in[2];
    const float* bb    = (const float*)d_in[3];
    const float* wc    = (const float*)d_in[4];
    const float* bc    = (const float*)d_in[5];
    const float* wd    = (const float*)d_in[6];
    const float* bd    = (const float*)d_in[7];
    const float* alpha = (const float*)d_in[8];
    float* out = (float*)d_out;

    char* ws = (char*)d_ws;
    unsigned short* Qbf   = (unsigned short*)(ws);                         // 2 MB
    unsigned short* Kbf   = (unsigned short*)(ws + (size_t)(1u << 21));    // 2 MB
    unsigned short* Vbf   = (unsigned short*)(ws + (size_t)(1u << 22));    // 16 MB
    unsigned short* Wb_bf = (unsigned short*)(ws + (size_t)(1u << 22) + (size_t)(16u << 20));
    unsigned short* Wc_bf = Wb_bf + CQK * CIN;
    unsigned short* Wd_bf = Wc_bf + CQK * CIN;

    // 1) weight conversion: 2*32768 + 262144 = 327680 elems
    k_prep<<<dim3(1280), dim3(256), 0, stream>>>(wb, wc, wd, Wb_bf, Wc_bf, Wd_bf);

    // 2) fused QKV projection: one block per (batch, 16-pixel tile)
    k_qkv<<<dim3(BATCH * (NPIX / 16)), dim3(256), 0, stream>>>(
        x, lf, Wb_bf, bb, Wc_bf, bc, Wd_bf, bd, Qbf, Kbf, Vbf);

    // 3) flash attention: one block per (batch, 16-query tile); ~257.5 KB LDS
    const size_t lds_bytes = (size_t)(16 * SROW) * 4 + 256 * 4 + 32 * 4;
    k_attn<<<dim3(BATCH * (NPIX / 16)), dim3(256), lds_bytes, stream>>>(
        Qbf, Kbf, Vbf, x, alpha, out);
}
